// GATClassifier_19078244729543
// MI455X (gfx1250) — matrix operand order repaired and checked by reading, not compiled, not run on hardware
//
#include <hip/hip_runtime.h>
#include <hip/hip_bf16.h>

// ---------------- types ----------------
typedef __bf16 v16bf __attribute__((ext_vector_type(16)));
typedef __bf16 v8bf  __attribute__((ext_vector_type(8)));
typedef float  v8f   __attribute__((ext_vector_type(8)));
typedef int    v4i_t __attribute__((ext_vector_type(4)));

#define N_NODES 20000
#define N_EDGES 320000
#define ET      (N_EDGES + N_NODES)   // + self loops
#define D_IN    768
#define HC1     1024                  // 8 heads * 128
#define H1      8
#define C1      128
#define HC2     128                   // 1 head * 128
#define H2      1
#define C2      128
#define NEG_SLOPE 0.2f
#define BN_EPS  1e-5f

// ---- gfx1250 async global->LDS path (guarded) ----
#if defined(__has_builtin)
# if __has_builtin(__builtin_amdgcn_global_load_async_to_lds_b128)
#  define HAVE_ASYNC_LDS 1
# endif
#endif
#ifndef HAVE_ASYNC_LDS
# define HAVE_ASYNC_LDS 0
#endif

#if defined(__has_builtin)
# if __has_builtin(__builtin_amdgcn_s_wait_asynccnt)
#  define WAIT_ASYNC(n) __builtin_amdgcn_s_wait_asynccnt(n)
# endif
#endif
#ifndef WAIT_ASYNC
# define WAIT_ASYNC(n) asm volatile("s_wait_asynccnt %0" :: "i"(n) : "memory")
#endif

#if HAVE_ASYNC_LDS
#define ASYNC_COPY16(SRC, DST)                                              \
    __builtin_amdgcn_global_load_async_to_lds_b128(                         \
        (__attribute__((address_space(1))) v4i_t*)(SRC),                    \
        (__attribute__((address_space(3))) v4i_t*)(DST), 0, 0)
#endif

// ---------------- helpers ----------------
__device__ __forceinline__ unsigned f2ord(float f) {
    unsigned u = __float_as_uint(f);
    return (u & 0x80000000u) ? ~u : (u | 0x80000000u);
}
__device__ __forceinline__ float ord2f(unsigned u) {
    return (u & 0x80000000u) ? __uint_as_float(u ^ 0x80000000u) : __uint_as_float(~u);
}

// ---------------- conversion kernels ----------------
__global__ void cvt_f32_bf16(const float* __restrict__ in, __bf16* __restrict__ out, size_t n) {
    size_t i = (size_t)blockIdx.x * blockDim.x + threadIdx.x;
    if (i < n) out[i] = (__bf16)in[i];
}

// W[K][Nd] (row-major) -> Wt[Nd][K] bf16
__global__ void transpose_cvt(const float* __restrict__ W, __bf16* __restrict__ Wt, int K, int Nd) {
    int i = blockIdx.x * blockDim.x + threadIdx.x;
    if (i >= K * Nd) return;
    int k = i / Nd, n = i % Nd;
    Wt[(size_t)n * K + k] = (__bf16)W[i];
}

// ---------------- WMMA GEMM: C[M][Nd] = A[M][K] * Bt[Nd][K]^T ----------------
// Block = 256 threads = 8 waves; block tile = 128 rows x 128 cols.
// B k-slice (128 cols x 32 k, bf16, row padded to 40 elems) staged in LDS,
// double-buffered via async global->LDS copies; shared by all 8 waves.
// All 8 B fragments are pulled into registers before issuing the 8 WMMAs.
#define BROW 40   // padded LDS row stride (elements)

__global__ void wmma_gemm_bf16(const __bf16* __restrict__ A,
                               const __bf16* __restrict__ Bt,
                               float* __restrict__ C,
                               int M, int Nd, int K) {
    __shared__ __bf16 Bs[2][128 * BROW];

    const int wave = threadIdx.x >> 5;
    const int lane = threadIdx.x & 31;
    const int tm = (blockIdx.x * 8 + wave) * 16;
    const int tn = blockIdx.y * 128;
    const bool active = tm < M;

    const int mrow = lane & 15;          // row (A) / col (B,C) index within tile
    const int half = lane >> 4;
    const int KT = K / 32;

    v8f acc[8];
#pragma unroll
    for (int t = 0; t < 8; ++t) acc[t] = (v8f)(0.0f);

    const __bf16* arow = A + (size_t)((active ? tm : 0) + mrow) * K;

    // cooperative copy: 512 16B chunks; thread handles chunk tid and tid+256
    const int c0n  = threadIdx.x >> 2;          // chunk -> B column
    const int c0k  = threadIdx.x & 3;           // chunk -> k-part (8 elems)
    const int c1n  = (threadIdx.x + 256) >> 2;
    const int c1k  = c0k;

#if HAVE_ASYNC_LDS
    // prologue: fill buffer 0
    {
        const __bf16* s0 = Bt + (size_t)(tn + c0n) * K + 0 + c0k * 8;
        const __bf16* s1 = Bt + (size_t)(tn + c1n) * K + 0 + c1k * 8;
        ASYNC_COPY16(s0, &Bs[0][c0n * BROW + c0k * 8]);
        ASYNC_COPY16(s1, &Bs[0][c1n * BROW + c1k * 8]);
    }
#endif

    // A-fragment register double buffer
    v8bf alo = (v8bf)(__bf16)0.f, ahi = (v8bf)(__bf16)0.f;
    if (active) {
        alo = *(const v8bf*)(arow + half * 8);
        ahi = *(const v8bf*)(arow + half * 8 + 16);
    }

    for (int kt = 0; kt < KT; ++kt) {
        const int cur = kt & 1;
        const int k0  = kt * 32;

#if HAVE_ASYNC_LDS
        if (kt + 1 < KT) {
            const int kn = k0 + 32;
            const __bf16* s0 = Bt + (size_t)(tn + c0n) * K + kn + c0k * 8;
            const __bf16* s1 = Bt + (size_t)(tn + c1n) * K + kn + c1k * 8;
            ASYNC_COPY16(s0, &Bs[cur ^ 1][c0n * BROW + c0k * 8]);
            ASYNC_COPY16(s1, &Bs[cur ^ 1][c1n * BROW + c1k * 8]);
            WAIT_ASYNC(2);           // older (current-buffer) copies complete
        } else {
            WAIT_ASYNC(0);
        }
        __syncthreads();
        const __bf16* bbase = &Bs[cur][0];
#else
        // synchronous fallback: single buffer
        uint4 r0, r1;
        {
            const __bf16* s0 = Bt + (size_t)(tn + c0n) * K + k0 + c0k * 8;
            const __bf16* s1 = Bt + (size_t)(tn + c1n) * K + k0 + c1k * 8;
            r0 = *(const uint4*)s0;
            r1 = *(const uint4*)s1;
        }
        __syncthreads();             // previous compute done reading LDS
        *(uint4*)&Bs[0][c0n * BROW + c0k * 8] = r0;
        *(uint4*)&Bs[0][c1n * BROW + c1k * 8] = r1;
        __syncthreads();
        const __bf16* bbase = &Bs[0][0];
#endif

        // prefetch next A fragment
        v8bf nlo = alo, nhi = ahi;
        if (active && kt + 1 < KT) {
            nlo = *(const v8bf*)(arow + k0 + 32 + half * 8);
            nhi = *(const v8bf*)(arow + k0 + 32 + half * 8 + 16);
        }

        if (active) {
            v16bf af;
#pragma unroll
            for (int j = 0; j < 8; ++j) { af[j] = alo[j]; af[8 + j] = ahi[j]; }

            // pull ALL 8 B fragments into registers first ...
            v8bf b0[8], b1[8];
#pragma unroll
            for (int t = 0; t < 8; ++t) {
                const __bf16* bp = bbase + (size_t)(t * 16 + mrow) * BROW + half * 8;
                b0[t] = *(const v8bf*)(bp);
                b1[t] = *(const v8bf*)(bp + 16);
            }
            // ... then issue 8 WMMAs back-to-back
#pragma unroll
            for (int t = 0; t < 8; ++t) {
                v16bf bf;
#pragma unroll
                for (int j = 0; j < 8; ++j) { bf[j] = b0[t][j]; bf[8 + j] = b1[t][j]; }
                acc[t] = __builtin_amdgcn_wmma_f32_16x16x32_bf16(
                    false, af, false, bf, (short)0, acc[t], false, false);
            }
        }
        alo = nlo; ahi = nhi;
        __syncthreads();             // all waves done reading before buffer reuse
    }

    if (active) {
        // D layout: VGPR r, lane L -> row = r + 8*half, col = L%16
#pragma unroll
        for (int t = 0; t < 8; ++t) {
            const int n = tn + t * 16 + mrow;
#pragma unroll
            for (int r = 0; r < 8; ++r) {
                const int m = tm + half * 8 + r;
                C[(size_t)m * Nd + n] = acc[t][r];
            }
        }
    }
}

// ---------------- attention scores: a_s/a_d per (node, head) ----------------
__global__ void att_scores(const float* __restrict__ h,
                           const float* __restrict__ att_s, const float* __restrict__ att_d,
                           int n, int H, int C,
                           float* __restrict__ a_s, float* __restrict__ a_d) {
    int i = blockIdx.x * blockDim.x + threadIdx.x;   // node*H + head
    if (i >= n * H) return;
    int hh = i % H;
    const float* row = h + (size_t)i * C;
    float s = 0.f, d = 0.f;
    for (int c = 0; c < C; ++c) {
        float v = row[c];
        s += v * att_s[hh * C + c];
        d += v * att_d[hh * C + c];
    }
    a_s[i] = s; a_d[i] = d;
}

// ---------------- edge kernels ----------------
__global__ void edge_logits(const long long* __restrict__ ei, int H,
                            const float* __restrict__ a_s, const float* __restrict__ a_d,
                            float* __restrict__ ebuf, unsigned* __restrict__ mbuf) {
    int idx = blockIdx.x * blockDim.x + threadIdx.x;
    if (idx >= ET * H) return;
    int e = idx / H, hh = idx % H;
    int s, d;
    if (e < N_EDGES) { s = (int)ei[e]; d = (int)ei[N_EDGES + e]; }
    else             { s = d = e - N_EDGES; }
    float v = a_s[s * H + hh] + a_d[d * H + hh];
    v = v > 0.f ? v : NEG_SLOPE * v;                   // leaky relu
    ebuf[idx] = v;
    atomicMax(mbuf + d * H + hh, f2ord(v));
}

__global__ void edge_expz(const long long* __restrict__ ei, int H,
                          float* __restrict__ ebuf, const unsigned* __restrict__ mbuf,
                          float* __restrict__ z) {
    int idx = blockIdx.x * blockDim.x + threadIdx.x;
    if (idx >= ET * H) return;
    int e = idx / H, hh = idx % H;
    int d = (e < N_EDGES) ? (int)ei[N_EDGES + e] : e - N_EDGES;
    float m  = ord2f(mbuf[d * H + hh]);
    float ex = __expf(ebuf[idx] - m);
    ebuf[idx] = ex;
    atomicAdd(z + d * H + hh, ex);
}

// one block per edge; scatter alpha * h[src] into out[dst]
__global__ void edge_aggregate(const long long* __restrict__ ei, int H, int C,
                               const float* __restrict__ exbuf, const float* __restrict__ z,
                               const float* __restrict__ h, float* __restrict__ out) {
    int e = blockIdx.x;
    int s, d;
    if (e < N_EDGES) { s = (int)ei[e]; d = (int)ei[N_EDGES + e]; }
    else             { s = d = e - N_EDGES; }
    const int HCd = H * C;
    for (int i = threadIdx.x; i < HCd; i += blockDim.x) {
        int hh = i / C;
        float alpha = exbuf[e * H + hh] / z[d * H + hh];
        atomicAdd(out + (size_t)d * HCd + i, alpha * h[(size_t)s * HCd + i]);
    }
}

// ---------------- batch norm ----------------
__global__ void bn_partial(const float* __restrict__ X, const float* __restrict__ bias,
                           int n, int Cd, int rows_per,
                           float* __restrict__ sacc, float* __restrict__ ssacc) {
    int col = blockIdx.x * blockDim.x + threadIdx.x;
    if (col >= Cd) return;
    int r0 = blockIdx.y * rows_per;
    int r1 = r0 + rows_per; if (r1 > n) r1 = n;
    float b = bias[col];
    float s = 0.f, ss = 0.f;
    for (int r = r0; r < r1; ++r) {
        float v = X[(size_t)r * Cd + col] + b;
        s += v; ss += v * v;
    }
    atomicAdd(&sacc[col], s);
    atomicAdd(&ssacc[col], ss);
}

__global__ void bn_finalize(const float* __restrict__ sacc, const float* __restrict__ ssacc,
                            int n, int Cd, float* __restrict__ mu, float* __restrict__ rsig) {
    int col = blockIdx.x * blockDim.x + threadIdx.x;
    if (col >= Cd) return;
    float m = sacc[col] / (float)n;
    float var = ssacc[col] / (float)n - m * m;
    mu[col] = m;
    rsig[col] = rsqrtf(var + BN_EPS);
}

// y = elu((x + bias - mu) * rsig * g + be); writes f32 and/or bf16
__global__ void bn_elu(const float* __restrict__ X, const float* __restrict__ bias,
                       const float* __restrict__ mu, const float* __restrict__ rsig,
                       const float* __restrict__ g, const float* __restrict__ be,
                       int n, int Cd,
                       float* __restrict__ Yf, __bf16* __restrict__ Ybf) {
    size_t i = (size_t)blockIdx.x * blockDim.x + threadIdx.x;
    if (i >= (size_t)n * Cd) return;
    int col = (int)(i % Cd);
    float v = (X[i] + bias[col] - mu[col]) * rsig[col] * g[col] + be[col];
    v = v > 0.f ? v : expm1f(v);
    if (Yf)  Yf[i]  = v;
    if (Ybf) Ybf[i] = (__bf16)v;
}

// ---------------- MLP head: relu(x@Wc1+bc1)@Wc2+bc2 ----------------
__global__ void mlp_head(const float* __restrict__ Xa,
                         const float* __restrict__ Wc1, const float* __restrict__ bc1,
                         const float* __restrict__ Wc2, const float* __restrict__ bc2,
                         int n, float* __restrict__ out) {
    int r = blockIdx.x * blockDim.x + threadIdx.x;
    if (r >= n) return;
    const float* x = Xa + (size_t)r * C2;
    float o0 = bc2[0], o1 = bc2[1];
    for (int j = 0; j < 64; ++j) {
        float s = bc1[j];
        for (int k = 0; k < C2; ++k) s += x[k] * Wc1[k * 64 + j];
        s = s > 0.f ? s : 0.f;
        o0 += s * Wc2[j * 2 + 0];
        o1 += s * Wc2[j * 2 + 1];
    }
    out[r * 2 + 0] = o0;
    out[r * 2 + 1] = o1;
}

// ---------------- launch ----------------
extern "C" void kernel_launch(void* const* d_in, const int* in_sizes, int n_in,
                              void* d_out, int out_size, void* d_ws, size_t ws_size,
                              hipStream_t stream) {
    const float*     x        = (const float*)d_in[0];
    const long long* ei       = (const long long*)d_in[1];   // int64 (2,E)
    const float*     W1       = (const float*)d_in[2];
    const float*     att_s1   = (const float*)d_in[3];
    const float*     att_d1   = (const float*)d_in[4];
    const float*     b1       = (const float*)d_in[5];
    const float*     W2       = (const float*)d_in[6];
    const float*     att_s2   = (const float*)d_in[7];
    const float*     att_d2   = (const float*)d_in[8];
    const float*     b2       = (const float*)d_in[9];
    const float*     g1       = (const float*)d_in[10];
    const float*     be1      = (const float*)d_in[11];
    const float*     g2       = (const float*)d_in[12];
    const float*     be2      = (const float*)d_in[13];
    const float*     Wc1      = (const float*)d_in[14];
    const float*     bc1      = (const float*)d_in[15];
    const float*     Wc2      = (const float*)d_in[16];
    const float*     bc2      = (const float*)d_in[17];
    float* out = (float*)d_out;

    // ---- workspace arena ----
    char* base = (char*)d_ws;
    size_t off = 0;
    auto alloc = [&](size_t bytes) -> char* {
        char* p = base + off;
        off = (off + bytes + 255) & ~(size_t)255;
        return p;
    };
    __bf16*  x_bf   = (__bf16*)alloc((size_t)N_NODES * D_IN * 2);
    __bf16*  W1t    = (__bf16*)alloc((size_t)HC1 * D_IN * 2);
    __bf16*  W2t    = (__bf16*)alloc((size_t)HC2 * HC1 * 2);
    char*    bufA   = alloc((size_t)N_NODES * HC1 * 4);  // h1 f32, later h1act bf16
    char*    bufB   = alloc((size_t)N_NODES * HC1 * 4);  // out1 f32, later h2 f32
    float*   a_s1b  = (float*)alloc((size_t)N_NODES * H1 * 4);
    float*   a_d1b  = (float*)alloc((size_t)N_NODES * H1 * 4);
    unsigned* m1    = (unsigned*)alloc((size_t)N_NODES * H1 * 4);
    float*   z1     = (float*)alloc((size_t)N_NODES * H1 * 4);
    float*   e1     = (float*)alloc((size_t)ET * H1 * 4);
    float*   a_s2b  = (float*)alloc((size_t)N_NODES * 4);
    float*   a_d2b  = (float*)alloc((size_t)N_NODES * 4);
    unsigned* m2    = (unsigned*)alloc((size_t)N_NODES * 4);
    float*   z2     = (float*)alloc((size_t)N_NODES * 4);
    float*   e2     = (float*)alloc((size_t)ET * 4);
    float*   out2   = (float*)alloc((size_t)N_NODES * HC2 * 4);
    float*   h2act  = (float*)alloc((size_t)N_NODES * HC2 * 4);
    float*   sacc   = (float*)alloc(1024 * 4);
    float*   ssacc  = (float*)alloc(1024 * 4);
    float*   mu1    = (float*)alloc(1024 * 4);
    float*   rsig1  = (float*)alloc(1024 * 4);
    float*   mu2    = (float*)alloc(128 * 4);
    float*   rsig2  = (float*)alloc(128 * 4);

    float*  h1     = (float*)bufA;
    float*  out1   = (float*)bufB;
    __bf16* h1act  = (__bf16*)bufA;   // reuse after h1 dead
    float*  h2     = (float*)bufB;    // reuse after out1 dead

    const int TPB = 256;

    // 1) convert inputs to bf16
    {
        size_t n = (size_t)N_NODES * D_IN;
        cvt_f32_bf16<<<(unsigned)((n + TPB - 1) / TPB), TPB, 0, stream>>>(x, x_bf, n);
        int n1 = D_IN * HC1;
        transpose_cvt<<<(n1 + TPB - 1) / TPB, TPB, 0, stream>>>(W1, W1t, D_IN, HC1);
        int n2 = HC1 * HC2;
        transpose_cvt<<<(n2 + TPB - 1) / TPB, TPB, 0, stream>>>(W2, W2t, HC1, HC2);
    }

    // 2) GEMM1: h1 = x @ W1   (M=20000, Nd=1024, K=768)
    {
        dim3 grid((N_NODES / 16 + 7) / 8, HC1 / 128);
        wmma_gemm_bf16<<<grid, TPB, 0, stream>>>(x_bf, W1t, h1, N_NODES, HC1, D_IN);
    }

    // 3) attention scores layer 1
    att_scores<<<(N_NODES * H1 + TPB - 1) / TPB, TPB, 0, stream>>>(
        h1, att_s1, att_d1, N_NODES, H1, C1, a_s1b, a_d1b);

    // 4) edge softmax + aggregate layer 1
    (void)hipMemsetAsync(m1, 0, (size_t)N_NODES * H1 * 4, stream);   // ord(0) < any encoded value
    (void)hipMemsetAsync(z1, 0, (size_t)N_NODES * H1 * 4, stream);
    edge_logits<<<(ET * H1 + TPB - 1) / TPB, TPB, 0, stream>>>(ei, H1, a_s1b, a_d1b, e1, m1);
    edge_expz<<<(ET * H1 + TPB - 1) / TPB, TPB, 0, stream>>>(ei, H1, e1, m1, z1);
    (void)hipMemsetAsync(out1, 0, (size_t)N_NODES * HC1 * 4, stream);
    edge_aggregate<<<ET, TPB, 0, stream>>>(ei, H1, C1, e1, z1, h1, out1);

    // 5) BN1 + ELU (bias folded in); emit bf16 for GEMM2
    (void)hipMemsetAsync(sacc, 0, 1024 * 4, stream);
    (void)hipMemsetAsync(ssacc, 0, 1024 * 4, stream);
    {
        dim3 g((HC1 + 127) / 128, 40);
        bn_partial<<<g, 128, 0, stream>>>(out1, b1, N_NODES, HC1, 512, sacc, ssacc);
        bn_finalize<<<(HC1 + 127) / 128, 128, 0, stream>>>(sacc, ssacc, N_NODES, HC1, mu1, rsig1);
        size_t n = (size_t)N_NODES * HC1;
        bn_elu<<<(unsigned)((n + TPB - 1) / TPB), TPB, 0, stream>>>(
            out1, b1, mu1, rsig1, g1, be1, N_NODES, HC1, (float*)nullptr, h1act);
    }

    // 6) GEMM2: h2 = h1act @ W2   (M=20000, Nd=128, K=1024)
    {
        dim3 grid((N_NODES / 16 + 7) / 8, HC2 / 128);
        wmma_gemm_bf16<<<grid, TPB, 0, stream>>>(h1act, W2t, h2, N_NODES, HC2, HC1);
    }

    // 7) attention scores layer 2
    att_scores<<<(N_NODES * H2 + TPB - 1) / TPB, TPB, 0, stream>>>(
        h2, att_s2, att_d2, N_NODES, H2, C2, a_s2b, a_d2b);

    // 8) edge softmax + aggregate layer 2
    (void)hipMemsetAsync(m2, 0, (size_t)N_NODES * 4, stream);
    (void)hipMemsetAsync(z2, 0, (size_t)N_NODES * 4, stream);
    edge_logits<<<(ET * H2 + TPB - 1) / TPB, TPB, 0, stream>>>(ei, H2, a_s2b, a_d2b, e2, m2);
    edge_expz<<<(ET * H2 + TPB - 1) / TPB, TPB, 0, stream>>>(ei, H2, e2, m2, z2);
    (void)hipMemsetAsync(out2, 0, (size_t)N_NODES * HC2 * 4, stream);
    edge_aggregate<<<ET, 128, 0, stream>>>(ei, H2, C2, e2, z2, h2, out2);

    // 9) BN2 + ELU -> h2act (f32)
    (void)hipMemsetAsync(sacc, 0, 128 * 4, stream);
    (void)hipMemsetAsync(ssacc, 0, 128 * 4, stream);
    {
        dim3 g(1, 40);
        bn_partial<<<g, 128, 0, stream>>>(out2, b2, N_NODES, HC2, 512, sacc, ssacc);
        bn_finalize<<<1, 128, 0, stream>>>(sacc, ssacc, N_NODES, HC2, mu2, rsig2);
        size_t n = (size_t)N_NODES * HC2;
        bn_elu<<<(unsigned)((n + TPB - 1) / TPB), TPB, 0, stream>>>(
            out2, b2, mu2, rsig2, g2, be2, N_NODES, HC2, h2act, (__bf16*)nullptr);
    }

    // 10) MLP head -> d_out [N,2]
    mlp_head<<<(N_NODES + TPB - 1) / TPB, TPB, 0, stream>>>(h2act, Wc1, bc1, Wc2, bc2, N_NODES, out);
}
